// GAT_Module_59631325937826
// MI455X (gfx1250) — matrix operand order repaired and checked by reading, not compiled, not run on hardware
//
#include <hip/hip_runtime.h>
#include <hip/hip_bf16.h>

// Problem constants (match reference)
#define NN 50000
#define EE 800000
#define HH 4
#define FF 64
#define HF 256               // H*F
#define NEG_SLOPE 0.2f

typedef __attribute__((ext_vector_type(16))) __bf16 v16bf;
typedef __attribute__((ext_vector_type(8)))  float  v8f;
typedef unsigned int u32x4 __attribute__((ext_vector_type(4)));
typedef int          i32x8 __attribute__((ext_vector_type(8)));
typedef int          i32x4 __attribute__((ext_vector_type(4)));

// ---- helpers -------------------------------------------------------------

// Monotonic float <-> uint encoding so atomicMax(u32) implements float max.
__device__ __forceinline__ unsigned enc_f32(float f) {
    unsigned u = __float_as_uint(f);
    return (u & 0x80000000u) ? ~u : (u | 0x80000000u);
}
__device__ __forceinline__ float dec_f32(unsigned e) {
    unsigned u = (e & 0x80000000u) ? (e ^ 0x80000000u) : ~e;
    return __uint_as_float(u);
}
__device__ __forceinline__ float leaky(float s) {
    return s > 0.f ? s : NEG_SLOPE * s;
}

// LDS byte offset of a __shared__ object (addrspacecast generic -> AS3).
__device__ __forceinline__ unsigned lds_off(const void* p) {
    return (unsigned)(unsigned long long)(__attribute__((address_space(3))) const char*)p;
}

// Issue a TDM 2-D tile load (tensor == tile; dims in 4-byte elements).
// D# layout per CDNA5 ISA; groups 2/3 zero.
// This toolchain exposes the 6-arg builtin: (g0, g1, g2, g3, g4, cpol).
__device__ __forceinline__ void tdm_load_2d_f32(unsigned lds_byte_off,
                                                const void* gaddr,
                                                unsigned d0, unsigned d1)
{
    unsigned long long ga = (unsigned long long)gaddr;
    u32x4 g0;
    g0.x = 0x1u;                                              // count=1, user D#
    g0.y = lds_byte_off;                                      // lds_addr [63:32]
    g0.z = (unsigned)(ga & 0xFFFFFFFFu);                      // global_addr[31:0]
    g0.w = (unsigned)((ga >> 32) & 0x01FFFFFFu) | 0x80000000u; // addr[56:32] | type=2
    i32x8 g1;
    g1[0] = 0x00020000;                       // wg_mask=0, data_size=2 (4B)
    g1[1] = (int)(d0 << 16);                  // tensor_dim0[15:0]  (bits 63:48)
    g1[2] = (int)((d0 >> 16) | (d1 << 16));   // tdim0 hi | tensor_dim1 lo
    g1[3] = (int)((d1 >> 16) | (d0 << 16));   // tdim1 hi | tile_dim0 (127:112)
    g1[4] = (int)(d1 & 0xFFFFu);              // tile_dim1 (143:128), tile_dim2=0
    g1[5] = (int)d0;                          // tensor_dim0_stride[31:0]
    g1[6] = 0;
    g1[7] = 0;
    i32x4 z4 = {0, 0, 0, 0};
    i32x8 z8 = {0, 0, 0, 0, 0, 0, 0, 0};
    __builtin_amdgcn_tensor_load_to_lds(g0, g1, z4, z4, z8, 0);
}

// ---- kernel 1: h = x @ W via v_wmma_f32_16x16x32_bf16 --------------------
// One block = 16 output rows x all 256 cols. TDM stages the A-tile (16x64)
// and the whole W (64x256) into LDS as f32; fragments convert to bf16.
__global__ __launch_bounds__(256)
void gat_gemm_wmma(const float* __restrict__ X,    // [NN, 64]
                   const float* __restrict__ W,    // [64, 256]
                   float* __restrict__ Hout)       // [NN, 256]
{
    __shared__ float ldsA[16 * 64];     // A tile row-major [m][k]   (4 KB)
    __shared__ float ldsW[64 * 256];    // W    row-major [k][n]     (64 KB)

    const int t    = threadIdx.x;
    const int tile = blockIdx.x;        // NN/16 = 3125 exact

    if (t < 32) {                       // one wave drives the TDM
        tdm_load_2d_f32(lds_off(ldsA), X + (size_t)tile * 16 * FF, FF, 16);
        tdm_load_2d_f32(lds_off(ldsW), W, HF, FF);
        __builtin_amdgcn_s_wait_tensorcnt(0);
    }
    __syncthreads();

    const int lane = t & 31;
    const int wave = t >> 5;            // 0..7
    const int half = lane >> 4;         // 0/1 (lane group)
    const int nlo  = lane & 15;

    // A fragments (16x32 bf16 layout, ISA 7.12.2): per lane, row m = nlo,
    // chunk0 = K[k0+half*8 .. +7], chunk1 = K[k0+16+half*8 .. +7]
    v16bf afrag[2];
    #pragma unroll
    for (int kk = 0; kk < 2; ++kk) {
        const int k0 = kk * 32;
        v16bf a;
        #pragma unroll
        for (int i = 0; i < 8; ++i) {
            a[i]     = (__bf16)ldsA[nlo * FF + k0 + half * 8 + i];
            a[i + 8] = (__bf16)ldsA[nlo * FF + k0 + 16 + half * 8 + i];
        }
        afrag[kk] = a;
    }

    #pragma unroll
    for (int j = 0; j < 2; ++j) {
        const int nt  = wave * 2 + j;       // N-tile 0..15
        const int col = nt * 16 + nlo;
        v8f acc = {};
        #pragma unroll
        for (int kk = 0; kk < 2; ++kk) {
            const int k0 = kk * 32;
            // B fragment (32x16 bf16): lane column n, K = k0+half*8 .. +7, k0+16+half*8 .. +7
            v16bf b;
            #pragma unroll
            for (int i = 0; i < 16; ++i)
                b[i] = (__bf16)ldsW[(k0 + (i & 8) * 2 + half * 8 + (i & 7)) * HF + col];
            acc = __builtin_amdgcn_wmma_f32_16x16x32_bf16(
                      false, afrag[kk], false, b, (short)0, acc, false, false);
        }
        // D layout: VGPR v -> row m = v + half*8, col = nt*16 + nlo
        #pragma unroll
        for (int v = 0; v < 8; ++v)
            Hout[(tile * 16 + v + half * 8) * HF + col] = acc[v];
    }
}

// ---- kernel 2: per-node attention logits el/er ---------------------------
__global__ __launch_bounds__(256)
void gat_node_logits(const float* __restrict__ Hfeat,   // [NN,256]
                     const float* __restrict__ al,      // [4,64]
                     const float* __restrict__ ar,      // [4,64]
                     float* __restrict__ el, float* __restrict__ er)
{
    int gid = blockIdx.x * 256 + threadIdx.x;
    if (gid >= NN * HH) return;
    int n = gid >> 2, h = gid & 3;
    const float* row = Hfeat + n * HF + h * FF;
    float sl = 0.f, sr = 0.f;
    #pragma unroll 8
    for (int f = 0; f < FF; ++f) {
        float v = row[f];
        sl += v * al[h * FF + f];
        sr += v * ar[h * FF + f];
    }
    el[gid] = sl;
    er[gid] = sr;
}

// ---- kernel 3: segment max of edge scores (atomicMax on encoded u32) -----
__global__ __launch_bounds__(256)
void gat_edge_max(const int* __restrict__ src, const int* __restrict__ dst,
                  const float* __restrict__ el, const float* __restrict__ er,
                  unsigned* __restrict__ emax)
{
    int gid = blockIdx.x * 256 + threadIdx.x;   // EE*HH = 3.2M exact
    int e = gid >> 2, h = gid & 3;
    int s = src[e], d = dst[e];
    float sc = leaky(el[s * HH + h] + er[d * HH + h]);
    atomicMax(&emax[d * HH + h], enc_f32(sc));
}

// ---- kernel 4: softmax denominator (recompute score, atomicAdd exp) ------
__global__ __launch_bounds__(256)
void gat_edge_denom(const int* __restrict__ src, const int* __restrict__ dst,
                    const float* __restrict__ el, const float* __restrict__ er,
                    const unsigned* __restrict__ emax, float* __restrict__ denom)
{
    int gid = blockIdx.x * 256 + threadIdx.x;
    int e = gid >> 2, h = gid & 3;
    int s = src[e], d = dst[e];
    float sc = leaky(el[s * HH + h] + er[d * HH + h]);
    atomicAdd(&denom[d * HH + h], __expf(sc - dec_f32(emax[d * HH + h])));
}

// ---- kernel 5: weighted aggregation (wave per edge, 8 f32/lane) ----------
__global__ __launch_bounds__(256)
void gat_edge_agg(const int* __restrict__ src, const int* __restrict__ dst,
                  const float* __restrict__ el, const float* __restrict__ er,
                  const unsigned* __restrict__ emax, const float* __restrict__ denom,
                  const float* __restrict__ Hfeat, float* __restrict__ out)
{
    int e = blockIdx.x * 8 + (threadIdx.x >> 5);   // EE/8 = 100000 exact
    int lane = threadIdx.x & 31;
    int s = src[e], d = dst[e];
    int h = lane >> 3;                              // 8 lanes per head
    float sc    = leaky(el[s * HH + h] + er[d * HH + h]);
    float alpha = __expf(sc - dec_f32(emax[d * HH + h])) / denom[d * HH + h];
    const float4* hp = (const float4*)(Hfeat + s * HF + lane * 8);
    float4 a0 = hp[0], a1 = hp[1];
    float* op = out + d * HF + lane * 8;
    atomicAdd(op + 0, a0.x * alpha);
    atomicAdd(op + 1, a0.y * alpha);
    atomicAdd(op + 2, a0.z * alpha);
    atomicAdd(op + 3, a0.w * alpha);
    atomicAdd(op + 4, a1.x * alpha);
    atomicAdd(op + 5, a1.y * alpha);
    atomicAdd(op + 6, a1.z * alpha);
    atomicAdd(op + 7, a1.w * alpha);
}

// ---- epilogues: bias + (tanh) + head-mean --------------------------------
__global__ __launch_bounds__(256)
void gat_epi_tanh_mean(const float* __restrict__ agg, const float* __restrict__ bias,
                       float* __restrict__ xout)
{
    int gid = blockIdx.x * 256 + threadIdx.x;   // NN*FF = 3.2M exact
    int n = gid >> 6, f = gid & 63;
    float acc = 0.f;
    #pragma unroll
    for (int h = 0; h < HH; ++h)
        acc += tanhf(agg[n * HF + h * FF + f] + bias[h * FF + f]);
    xout[gid] = acc * 0.25f;
}

__global__ __launch_bounds__(256)
void gat_epi_mean(const float* __restrict__ agg, const float* __restrict__ bias,
                  float* __restrict__ yout)
{
    int gid = blockIdx.x * 256 + threadIdx.x;
    int n = gid >> 6, f = gid & 63;
    float acc = 0.f;
    #pragma unroll
    for (int h = 0; h < HH; ++h)
        acc += agg[n * HF + h * FF + f] + bias[h * FF + f];
    yout[gid] = acc * 0.25f;
}

// ---- host launcher -------------------------------------------------------
extern "C" void kernel_launch(void* const* d_in, const int* in_sizes, int n_in,
                              void* d_out, int out_size, void* d_ws, size_t ws_size,
                              hipStream_t stream) {
    const float* x   = (const float*)d_in[0];
    const int*   src = (const int*)  d_in[1];
    const int*   dst = (const int*)  d_in[2];
    const float* W1  = (const float*)d_in[3];
    const float* al1 = (const float*)d_in[4];
    const float* ar1 = (const float*)d_in[5];
    const float* b1  = (const float*)d_in[6];
    const float* W2  = (const float*)d_in[7];
    const float* al2 = (const float*)d_in[8];
    const float* ar2 = (const float*)d_in[9];
    const float* b2  = (const float*)d_in[10];
    (void)in_sizes; (void)n_in; (void)out_size; (void)ws_size;

    // Workspace layout (floats): 29.6M elems = 118.4 MB
    float*    ws    = (float*)d_ws;
    float*    Hfeat = ws;                       // NN*HF = 12.8M
    float*    agg   = Hfeat + (size_t)NN * HF;  // 12.8M
    float*    x2    = agg   + (size_t)NN * HF;  // NN*FF = 3.2M
    float*    el    = x2    + (size_t)NN * FF;  // NN*HH = 0.2M
    float*    er    = el    + (size_t)NN * HH;
    unsigned* emax  = (unsigned*)(er + (size_t)NN * HH);
    float*    denom = (float*)(emax + (size_t)NN * HH);

    const int gGemm  = NN / 16;                 // 3125
    const int gNH    = (NN * HH + 255) / 256;   // 782
    const int gEH    = (EE * HH) / 256;         // 12500
    const int gAgg   = EE / 8;                  // 100000
    const int gNF    = (NN * FF) / 256;         // 12500

    // ---------------- Layer 1 ----------------
    gat_gemm_wmma<<<gGemm, 256, 0, stream>>>(x, W1, Hfeat);
    gat_node_logits<<<gNH, 256, 0, stream>>>(Hfeat, al1, ar1, el, er);
    (void)hipMemsetAsync(emax,  0, (size_t)NN * HH * sizeof(unsigned), stream);
    (void)hipMemsetAsync(denom, 0, (size_t)NN * HH * sizeof(float),    stream);
    (void)hipMemsetAsync(agg,   0, (size_t)NN * HF * sizeof(float),    stream);
    gat_edge_max  <<<gEH,  256, 0, stream>>>(src, dst, el, er, emax);
    gat_edge_denom<<<gEH,  256, 0, stream>>>(src, dst, el, er, emax, denom);
    gat_edge_agg  <<<gAgg, 256, 0, stream>>>(src, dst, el, er, emax, denom, Hfeat, agg);
    gat_epi_tanh_mean<<<gNF, 256, 0, stream>>>(agg, b1, x2);

    // ---------------- Layer 2 ----------------
    gat_gemm_wmma<<<gGemm, 256, 0, stream>>>(x2, W2, Hfeat);
    gat_node_logits<<<gNH, 256, 0, stream>>>(Hfeat, al2, ar2, el, er);
    (void)hipMemsetAsync(emax,  0, (size_t)NN * HH * sizeof(unsigned), stream);
    (void)hipMemsetAsync(denom, 0, (size_t)NN * HH * sizeof(float),    stream);
    (void)hipMemsetAsync(agg,   0, (size_t)NN * HF * sizeof(float),    stream);
    gat_edge_max  <<<gEH,  256, 0, stream>>>(src, dst, el, er, emax);
    gat_edge_denom<<<gEH,  256, 0, stream>>>(src, dst, el, er, emax, denom);
    gat_edge_agg  <<<gAgg, 256, 0, stream>>>(src, dst, el, er, emax, denom, Hfeat, agg);
    gat_epi_mean<<<gNF, 256, 0, stream>>>(agg, b2, (float*)d_out);
}
